// EPCOR_6262062318063
// MI455X (gfx1250) — matrix operand order repaired and checked by reading, not compiled, not run on hardware
//
#include <hip/hip_runtime.h>

// EPCOR selectCom_adap + get_sparse_w for MI455X (gfx1250, wave32).
// pd [8,2048,2048] (134MB) is never materialized: each reduction pass
// recomputes pd tiles with bf16x3-split V_WMMA_F32_16X16X32_BF16 from
// pre-swizzled fragment-major bf16 hi/lo arrays (global_load_b128, zero
// per-tile cvt work). v3: two independent 16x16 tiles per loop iteration so
// the two WMMA accumulator chains interleave and each tile's VALU softmax
// epilogue co-executes in the other tile's XDL shadow (the ISA charges
// 4 coexec NOPs for WMMA->VALU D-reads; dual chains fill them).

namespace {
constexpr int B_ = 8;
constexpr int C_ = 64;
constexpr int N_ = 2048;
constexpr int M_ = 2048;
constexpr int T_ = N_ / 16;            // 128 tiles per sweep
constexpr int KSEL = 614;              // int(2048 * 0.3)
constexpr size_t SWZ = (size_t)B_ * T_ * 2 * 32 * 16;  // bf16 elems per swizzle array
}

typedef __attribute__((ext_vector_type(16))) __bf16 v16bf;
typedef __attribute__((ext_vector_type(8)))  float  v8f;

// ---------------- cross-lane helpers (16-lane groups) ----------------------
__device__ __forceinline__ float red16_add(float x) {
#pragma unroll
  for (int d = 1; d < 16; d <<= 1) x += __shfl_xor(x, d, 32);
  return x;
}

// lane-local online softmax accumulate
__device__ __forceinline__ void online_update(float& mx, float& sm, float pd) {
  const float nm = fmaxf(mx, pd);
  sm = sm * __expf(mx - nm) + __expf(pd - nm);
  mx = nm;
}
// merge (max,sum) pairs across the 16-lane group
__device__ __forceinline__ void online_merge16(float& mx, float& sm) {
#pragma unroll
  for (int d = 1; d < 16; d <<= 1) {
    const float om = __shfl_xor(mx, d, 32);
    const float os = __shfl_xor(sm, d, 32);
    const float nm = fmaxf(mx, om);
    sm = sm * __expf(mx - nm) + os * __expf(om - nm);
    mx = nm;
  }
}

// ---------------- A-fragment loader (contiguous-k 16-bit A layout, same as B) -----------
// A 16x32 (rows x k): lane L row = L&15; g=L>>4; slot j: k = kbase+g*16+j
__device__ __forceinline__ void loadA(const float* __restrict__ emb, int ld,
                                      int row0, int kbase, int lane,
                                      v16bf& hi, v16bf& lo) {
  const int row = row0 + (lane & 15);
  const int g = (lane >> 4) & 1;
#pragma unroll
  for (int j = 0; j < 16; ++j) {
    const int k = kbase + g * 16 + j;
    const float x = emb[k * ld + row];
    const __bf16 h = (__bf16)x;
    hi[j] = h;
    lo[j] = (__bf16)(x - (float)h);
  }
}

// bf16x3 product: hi*hi + hi*lo + lo*hi accumulated in f32.
__device__ __forceinline__ v8f wmma3(const v16bf& ah, const v16bf& al,
                                     const v16bf& bh, const v16bf& bl, v8f acc) {
  acc = __builtin_amdgcn_wmma_f32_16x16x32_bf16(false, ah, false, bh, (short)0, acc, false, false);
  acc = __builtin_amdgcn_wmma_f32_16x16x32_bf16(false, ah, false, bl, (short)0, acc, false, false);
  acc = __builtin_amdgcn_wmma_f32_16x16x32_bf16(false, al, false, bh, (short)0, acc, false, false);
  return acc;
}

// 16x16 inner tile from pre-swizzled B fragments. fragIdx = (b*T + t)*2.
__device__ __forceinline__ v8f inner_tile(const __bf16* __restrict__ hiB,
                                          const __bf16* __restrict__ loB,
                                          size_t fragIdx, int lane,
                                          const v16bf* ah, const v16bf* al) {
  const v16bf* H = (const v16bf*)hiB;
  const v16bf* L = (const v16bf*)loB;
  const v16bf bh0 = H[fragIdx * 32 + lane];
  const v16bf bl0 = L[fragIdx * 32 + lane];
  const v16bf bh1 = H[(fragIdx + 1) * 32 + lane];
  const v16bf bl1 = L[(fragIdx + 1) * 32 + lane];
  v8f acc = {};
  acc = wmma3(ah[0], al[0], bh0, bl0, acc);
  acc = wmma3(ah[1], al[1], bh1, bl1, acc);
  return acc;
}

__device__ __forceinline__ void prefetch_frag(const __bf16* __restrict__ hiB,
                                              size_t fragIdx, int lane) {
  const v16bf* H = (const v16bf*)hiB;
  __builtin_prefetch((const void*)(H + fragIdx * 32 + lane), 0, 1);
}

// ---------------- pre-swizzle f32 emb -> B-fragment-major bf16 hi/lo -------
// B 32x16 (k x col): lane L col = L&15; slot j: k = kbase + (L>>4)*16 + j.
__global__ __launch_bounds__(256) void k_swz(const float* __restrict__ emb, int ld,
                                             __bf16* __restrict__ hiA,
                                             __bf16* __restrict__ loA) {
  const int i = blockIdx.x * 256 + threadIdx.x;  // one thread per (b,t,kk,lane)
  const int tiles = ld / 16;
  if (i >= B_ * tiles * 2 * 32) return;
  const int lane = i & 31;
  const int kk = (i >> 5) & 1;
  const int t = (i >> 6) % tiles;
  const int b = i / (64 * tiles);
  const int col = t * 16 + (lane & 15);
  const int cbase = kk * 32 + ((lane >> 4) & 1) * 16;
  const float* e = emb + (size_t)b * C_ * ld + col;
  v16bf hv, lv;
#pragma unroll
  for (int j = 0; j < 16; ++j) {
    const float x = e[(cbase + j) * ld];
    const __bf16 h = (__bf16)x;
    hv[j] = h;
    lv[j] = (__bf16)(x - (float)h);
  }
  ((v16bf*)hiA)[i] = hv;
  ((v16bf*)loA)[i] = lv;
}

// ---------------- squared norms --------------------------------------------
__global__ __launch_bounds__(256) void k_sqnorm(const float* __restrict__ emb,
                                                float* __restrict__ out, int ld) {
  const int i = blockIdx.x * blockDim.x + threadIdx.x;
  if (i >= B_ * ld) return;
  const int b = i / ld, n = i % ld;
  const float* e = emb + (size_t)b * C_ * ld + n;
  float s = 0.f;
#pragma unroll 8
  for (int c = 0; c < C_; ++c) { const float v = e[c * ld]; s += v * v; }
  out[i] = s;
}

// ---------------- pass A: row softmax stats (axis=2) -----------------------
__global__ __launch_bounds__(128) void k_rowstats(const float* __restrict__ srcE,
                                                  const __bf16* __restrict__ tgtHi,
                                                  const __bf16* __restrict__ tgtLo,
                                                  const float* __restrict__ xx,
                                                  const float* __restrict__ yy,
                                                  float* __restrict__ rowMax,
                                                  float* __restrict__ rowZ) {
  const int b = blockIdx.y;
  const int wave = threadIdx.x >> 5, lane = threadIdx.x & 31;
  const int n0 = blockIdx.x * 64 + wave * 16;
  const float* sE = srcE + (size_t)b * C_ * N_;
  v16bf ah[2], al[2];
  loadA(sE, N_, n0, 0, lane, ah[0], al[0]);
  loadA(sE, N_, n0, 32, lane, ah[1], al[1]);
  const int g = (lane >> 4) & 1;
  float xv[8], rmax[8], rsum[8];
#pragma unroll
  for (int v = 0; v < 8; ++v) {
    xv[v] = xx[b * N_ + n0 + v + 8 * g];
    rmax[v] = -3.4e38f;
    rsum[v] = 0.f;
  }
  const size_t fb = (size_t)b * T_ * 2;
  const float* yyl = yy + b * M_ + (lane & 15);
  for (int t = 0; t < T_; t += 2) {
    const int tp = (t + 8 < T_) ? t + 8 : t;
    prefetch_frag(tgtHi, fb + 2 * tp, lane);
    const v8f acc0 = inner_tile(tgtHi, tgtLo, fb + 2 * t, lane, ah, al);
    const v8f acc1 = inner_tile(tgtHi, tgtLo, fb + 2 * (t + 1), lane, ah, al);
    const float yv0 = yyl[t * 16];
    const float yv1 = yyl[t * 16 + 16];
#pragma unroll
    for (int v = 0; v < 8; ++v)
      online_update(rmax[v], rsum[v], 2.f * acc0[v] - xv[v] - yv0);
#pragma unroll
    for (int v = 0; v < 8; ++v)
      online_update(rmax[v], rsum[v], 2.f * acc1[v] - xv[v] - yv1);
  }
#pragma unroll
  for (int v = 0; v < 8; ++v) {
    online_merge16(rmax[v], rsum[v]);
    if ((lane & 15) == 0) {
      const int row = n0 + v + 8 * g;
      rowMax[b * N_ + row] = rmax[v];
      rowZ[b * N_ + row] = rsum[v];
    }
  }
}

// ---------------- pass B: col softmax stats (axis=1) + scoresColSum --------
// Transposed ownership: WG owns m-rows, loops n (src acts as B-side).
__global__ __launch_bounds__(128) void k_colstats(const __bf16* __restrict__ srcHi,
                                                  const __bf16* __restrict__ srcLo,
                                                  const float* __restrict__ tgtE,
                                                  const float* __restrict__ xx,
                                                  const float* __restrict__ yy,
                                                  const float* __restrict__ rowMax,
                                                  const float* __restrict__ rowZ,
                                                  float* __restrict__ colMax,
                                                  float* __restrict__ colZ,
                                                  float* __restrict__ colSum) {
  const int b = blockIdx.y;
  const int wave = threadIdx.x >> 5, lane = threadIdx.x & 31;
  const int m0 = blockIdx.x * 64 + wave * 16;
  const float* tE = tgtE + (size_t)b * C_ * M_;
  v16bf ah[2], al[2];
  loadA(tE, M_, m0, 0, lane, ah[0], al[0]);
  loadA(tE, M_, m0, 32, lane, ah[1], al[1]);
  const int g = (lane >> 4) & 1;
  float yv[8], cmax[8], czum[8], scs[8];
#pragma unroll
  for (int v = 0; v < 8; ++v) {
    yv[v] = yy[b * M_ + m0 + v + 8 * g];
    cmax[v] = -3.4e38f;
    czum[v] = 0.f;
    scs[v] = 0.f;
  }
  const size_t fb = (size_t)b * T_ * 2;
  const float* xxl = xx + b * N_ + (lane & 15);
  const float* rMl = rowMax + b * N_ + (lane & 15);
  const float* rZl = rowZ + b * N_ + (lane & 15);
  for (int t = 0; t < T_; t += 2) {
    const int tp = (t + 8 < T_) ? t + 8 : t;
    prefetch_frag(srcHi, fb + 2 * tp, lane);
    const v8f acc0 = inner_tile(srcHi, srcLo, fb + 2 * t, lane, ah, al);
    const v8f acc1 = inner_tile(srcHi, srcLo, fb + 2 * (t + 1), lane, ah, al);
    const float xn0 = xxl[t * 16], xn1 = xxl[t * 16 + 16];
    const float rM0 = rMl[t * 16], rM1 = rMl[t * 16 + 16];
    const float rI0 = 1.f / rZl[t * 16], rI1 = 1.f / rZl[t * 16 + 16];
#pragma unroll
    for (int v = 0; v < 8; ++v) {
      const float pd = 2.f * acc0[v] - yv[v] - xn0;
      online_update(cmax[v], czum[v], pd);
      scs[v] += __expf(pd - rM0) * rI0;
    }
#pragma unroll
    for (int v = 0; v < 8; ++v) {
      const float pd = 2.f * acc1[v] - yv[v] - xn1;
      online_update(cmax[v], czum[v], pd);
      scs[v] += __expf(pd - rM1) * rI1;
    }
  }
#pragma unroll
  for (int v = 0; v < 8; ++v) {
    online_merge16(cmax[v], czum[v]);
    const float t = red16_add(scs[v]);
    if ((lane & 15) == 0) {
      const int m = m0 + v + 8 * g;
      colMax[b * M_ + m] = cmax[v];
      colZ[b * M_ + m] = czum[v];
      colSum[b * M_ + m] = t;
    }
  }
}

// ---------------- pass C: scoresRowSum (needs col stats) -------------------
__global__ __launch_bounds__(128) void k_rowsum(const float* __restrict__ srcE,
                                                const __bf16* __restrict__ tgtHi,
                                                const __bf16* __restrict__ tgtLo,
                                                const float* __restrict__ xx,
                                                const float* __restrict__ yy,
                                                const float* __restrict__ colMax,
                                                const float* __restrict__ colZ,
                                                float* __restrict__ rowSum) {
  const int b = blockIdx.y;
  const int wave = threadIdx.x >> 5, lane = threadIdx.x & 31;
  const int n0 = blockIdx.x * 64 + wave * 16;
  const float* sE = srcE + (size_t)b * C_ * N_;
  v16bf ah[2], al[2];
  loadA(sE, N_, n0, 0, lane, ah[0], al[0]);
  loadA(sE, N_, n0, 32, lane, ah[1], al[1]);
  const int g = (lane >> 4) & 1;
  float xv[8], rs[8];
#pragma unroll
  for (int v = 0; v < 8; ++v) {
    xv[v] = xx[b * N_ + n0 + v + 8 * g];
    rs[v] = 0.f;
  }
  const size_t fb = (size_t)b * T_ * 2;
  const float* yyl = yy + b * M_ + (lane & 15);
  const float* cMl = colMax + b * M_ + (lane & 15);
  const float* cZl = colZ + b * M_ + (lane & 15);
  for (int t = 0; t < T_; t += 2) {
    const int tp = (t + 8 < T_) ? t + 8 : t;
    prefetch_frag(tgtHi, fb + 2 * tp, lane);
    const v8f acc0 = inner_tile(tgtHi, tgtLo, fb + 2 * t, lane, ah, al);
    const v8f acc1 = inner_tile(tgtHi, tgtLo, fb + 2 * (t + 1), lane, ah, al);
    const float yv0 = yyl[t * 16], yv1 = yyl[t * 16 + 16];
    const float cM0 = cMl[t * 16], cM1 = cMl[t * 16 + 16];
    const float cI0 = 1.f / cZl[t * 16], cI1 = 1.f / cZl[t * 16 + 16];
#pragma unroll
    for (int v = 0; v < 8; ++v)
      rs[v] += __expf(2.f * acc0[v] - xv[v] - yv0 - cM0) * cI0;
#pragma unroll
    for (int v = 0; v < 8; ++v)
      rs[v] += __expf(2.f * acc1[v] - xv[v] - yv1 - cM1) * cI1;
  }
#pragma unroll
  for (int v = 0; v < 8; ++v) {
    const float t = red16_add(rs[v]);
    if ((lane & 15) == 0) rowSum[b * N_ + n0 + v + 8 * g] = t;
  }
}

// ---------------- k-th-smallest threshold + mask ---------------------------
// threshold = max{ v_i : #(v_j < v_i) <= k-1 } (== k-th smallest value)
__global__ __launch_bounds__(256) void k_select(const float* __restrict__ vals, int L,
                                                int ksel, float* __restrict__ maskWs,
                                                float* __restrict__ maskOut) {
  __shared__ float sv[2048];
  __shared__ float smax[256];
  const int b = blockIdx.x;
  const float* v = vals + b * L;
  for (int i = threadIdx.x; i < L; i += 256) sv[i] = v[i];
  __syncthreads();
  float best = -3.4e38f;
  for (int i = threadIdx.x; i < L; i += 256) {
    const float x = sv[i];
    int cnt = 0;
    for (int j = 0; j < L; ++j) cnt += (sv[j] < x) ? 1 : 0;
    if (cnt <= ksel - 1) best = fmaxf(best, x);
  }
  smax[threadIdx.x] = best;
  __syncthreads();
  for (int s = 128; s > 0; s >>= 1) {
    if (threadIdx.x < s) smax[threadIdx.x] = fmaxf(smax[threadIdx.x], smax[threadIdx.x + s]);
    __syncthreads();
  }
  const float thr = smax[0];
  for (int i = threadIdx.x; i < L; i += 256) {
    const float mk = (sv[i] < thr) ? 1.f : 0.f;
    maskWs[b * L + i] = mk;
    maskOut[b * L + i] = mk;
  }
}

// ---------------- pass D: sparse weights + src_corr + val_sum --------------
__global__ __launch_bounds__(128) void k_final(const float* __restrict__ srcE,
                                               const __bf16* __restrict__ tgtHi,
                                               const __bf16* __restrict__ tgtLo,
                                               const float* __restrict__ tgtPts,
                                               const float* __restrict__ xx,
                                               const float* __restrict__ yy,
                                               const float* __restrict__ rowMax,
                                               const float* __restrict__ rowZ,
                                               const float* __restrict__ maskS,
                                               const float* __restrict__ maskT,
                                               float* __restrict__ srcCorr,
                                               float* __restrict__ valsum) {
  const int b = blockIdx.y;
  const int wave = threadIdx.x >> 5, lane = threadIdx.x & 31;
  const int n0 = blockIdx.x * 64 + wave * 16;
  const float* sE = srcE + (size_t)b * C_ * N_;
  v16bf ah[2], al[2];
  loadA(sE, N_, n0, 0, lane, ah[0], al[0]);
  loadA(sE, N_, n0, 32, lane, ah[1], al[1]);
  const int g = (lane >> 4) & 1;
  float xv[8], rMv[8], rIv[8], msv[8], cs[8], a0[8], a1[8], a2[8];
#pragma unroll
  for (int v = 0; v < 8; ++v) {
    const int row = n0 + v + 8 * g;
    xv[v] = xx[b * N_ + row];
    rMv[v] = rowMax[b * N_ + row];
    rIv[v] = 1.f / rowZ[b * N_ + row];  // also == top-1 softmax value per row
    msv[v] = maskS[b * N_ + row];
    cs[v] = 0.f; a0[v] = 0.f; a1[v] = 0.f; a2[v] = 0.f;
  }
  const size_t fb = (size_t)b * T_ * 2;
  const float* yyl = yy + b * M_ + (lane & 15);
  const float* mtl = maskT + b * M_ + (lane & 15);
  const float* t0l = tgtPts + ((size_t)b * 3 + 0) * M_ + (lane & 15);
  const float* t1l = tgtPts + ((size_t)b * 3 + 1) * M_ + (lane & 15);
  const float* t2l = tgtPts + ((size_t)b * 3 + 2) * M_ + (lane & 15);
  for (int t = 0; t < T_; t += 2) {
    const int tp = (t + 8 < T_) ? t + 8 : t;
    prefetch_frag(tgtHi, fb + 2 * tp, lane);
    const v8f acc0 = inner_tile(tgtHi, tgtLo, fb + 2 * t, lane, ah, al);
    const v8f acc1 = inner_tile(tgtHi, tgtLo, fb + 2 * (t + 1), lane, ah, al);
#pragma unroll
    for (int u = 0; u < 2; ++u) {
      const int o = t * 16 + u * 16;
      const float yv = yyl[o];
      const float mt = mtl[o];
      const float t0 = t0l[o], t1 = t1l[o], t2 = t2l[o];
      const v8f& acc = u ? acc1 : acc0;
#pragma unroll
      for (int v = 0; v < 8; ++v) {
        const float pd = 2.f * acc[v] - xv[v] - yv;
        const float s = __expf(pd - rMv[v]) * rIv[v];  // scoresSoftCol
        const float maskf = ((msv[v] != 0.f) || (mt != 0.f)) ? 1.f : 0.f;
        // torch precedence: masked iff (maskf + s) < topk; topk == rIv
        const float w = ((maskf + s) < rIv[v]) ? 0.f : s;
        cs[v] += w;
        a0[v] += w * t0;
        a1[v] += w * t1;
        a2[v] += w * t2;
      }
    }
  }
#pragma unroll
  for (int v = 0; v < 8; ++v) {
    const float csr = red16_add(cs[v]);
    const float A0 = red16_add(a0[v]);
    const float A1 = red16_add(a1[v]);
    const float A2 = red16_add(a2[v]);
    if ((lane & 15) == 0) {
      const int row = n0 + v + 8 * g;
      const float cc = fmaxf(csr, 1e-5f);
      const float inv = 1.f / cc;
      srcCorr[((size_t)b * 3 + 0) * N_ + row] = A0 * inv;
      srcCorr[((size_t)b * 3 + 1) * N_ + row] = A1 * inv;
      srcCorr[((size_t)b * 3 + 2) * N_ + row] = A2 * inv;
      const float vs = csr * inv;
      valsum[b * N_ + row] = (msv[v] != 0.f) ? 0.f : vs;
    }
  }
}

// ---------------- src_weight: deterministic per-batch normalize ------------
__global__ __launch_bounds__(256) void k_weight(const float* __restrict__ valsum,
                                                float* __restrict__ srcW) {
  __shared__ float sh[256];
  const int b = blockIdx.x;
  const float* v = valsum + b * N_;
  float s = 0.f;
  for (int i = threadIdx.x; i < N_; i += 256) s += v[i];
  sh[threadIdx.x] = s;
  __syncthreads();
  for (int t = 128; t > 0; t >>= 1) {
    if (threadIdx.x < t) sh[threadIdx.x] += sh[threadIdx.x + t];
    __syncthreads();
  }
  const float inv = 1.f / sh[0];
  for (int i = threadIdx.x; i < N_; i += 256) srcW[b * N_ + i] = v[i] * inv;
}

// ---------------- launch ---------------------------------------------------
extern "C" void kernel_launch(void* const* d_in, const int* in_sizes, int n_in,
                              void* d_out, int out_size, void* d_ws, size_t ws_size,
                              hipStream_t stream) {
  (void)in_sizes; (void)n_in; (void)out_size; (void)ws_size;
  const float* srcE = (const float*)d_in[0];   // [B,C,N]
  const float* tgtE = (const float*)d_in[1];   // [B,C,M]
  const float* tgtP = (const float*)d_in[3];   // [B,3,M]  (d_in[2]=src unused)

  float* ws = (float*)d_ws;
  float* xx     = ws;
  float* yy     = ws + 1 * 16384;
  float* rowMax = ws + 2 * 16384;
  float* rowZ   = ws + 3 * 16384;
  float* colMax = ws + 4 * 16384;
  float* colZ   = ws + 5 * 16384;
  float* colSum = ws + 6 * 16384;
  float* rowSum = ws + 7 * 16384;
  float* maskS  = ws + 8 * 16384;
  float* maskT  = ws + 9 * 16384;
  float* valsum = ws + 10 * 16384;
  __bf16* tgtHi = (__bf16*)(ws + 11 * 16384);
  __bf16* tgtLo = tgtHi + SWZ;
  __bf16* srcHi = tgtLo + SWZ;
  __bf16* srcLo = srcHi + SWZ;

  float* out = (float*)d_out;
  float* srcCorr  = out;                        // [B,3,N]
  float* srcW     = out + (size_t)B_ * 3 * N_;  // [B,N]
  float* maskSOut = srcW + (size_t)B_ * N_;     // [B,N]
  float* maskTOut = maskSOut + (size_t)B_ * N_; // [B,M]

  const dim3 blk(128);
  const dim3 grd(N_ / 64, B_);
  const int swzThreads = B_ * T_ * 2 * 32;      // 65536

  k_swz<<<swzThreads / 256, 256, 0, stream>>>(tgtE, M_, tgtHi, tgtLo);
  k_swz<<<swzThreads / 256, 256, 0, stream>>>(srcE, N_, srcHi, srcLo);
  k_sqnorm<<<(B_ * N_) / 256, 256, 0, stream>>>(srcE, xx, N_);
  k_sqnorm<<<(B_ * M_) / 256, 256, 0, stream>>>(tgtE, yy, M_);
  k_rowstats<<<grd, blk, 0, stream>>>(srcE, tgtHi, tgtLo, xx, yy, rowMax, rowZ);
  k_colstats<<<grd, blk, 0, stream>>>(srcHi, srcLo, tgtE, xx, yy, rowMax, rowZ,
                                      colMax, colZ, colSum);
  k_select<<<B_, 256, 0, stream>>>(colSum, M_, KSEL, maskT, maskTOut);
  k_rowsum<<<grd, blk, 0, stream>>>(srcE, tgtHi, tgtLo, xx, yy, colMax, colZ, rowSum);
  k_select<<<B_, 256, 0, stream>>>(rowSum, N_, KSEL, maskS, maskSOut);
  k_final<<<grd, blk, 0, stream>>>(srcE, tgtHi, tgtLo, tgtP, xx, yy, rowMax, rowZ,
                                   maskS, maskT, srcCorr, valsum);
  k_weight<<<B_, 256, 0, stream>>>(valsum, srcW);
}
